// Model_80461917323570
// MI455X (gfx1250) — matrix operand order repaired and checked by reading, not compiled, not run on hardware
//
#include <hip/hip_runtime.h>
#include <hip/hip_bf16.h>
#include <math.h>

// Problem constants (match reference)
#define BB 4
#define TT 1024
#define DDIM 1024
#define HNUM 16
#define DK 64
#define TBD 64

typedef __attribute__((ext_vector_type(16))) __bf16 v16bf;
typedef __attribute__((ext_vector_type(8)))  __bf16 v8bf;
typedef __attribute__((ext_vector_type(8)))  float  v8f;
typedef __attribute__((ext_vector_type(4)))  int    v4i;
typedef __attribute__((address_space(1))) v4i v4i_as1;   // global
typedef __attribute__((address_space(3))) v4i v4i_as3;   // LDS

// ---------------------------------------------------------------------------
// Async global->LDS path (gfx1250): GLOBAL_LOAD_ASYNC_TO_LDS_B128 + ASYNCcnt
// Probe showed the builtin takes typed pointers: (int4 as1*, int4 as3*, imm, imm)
// ---------------------------------------------------------------------------
#if defined(__has_builtin)
#if __has_builtin(__builtin_amdgcn_global_load_async_to_lds_b128)
#define USE_ASYNC_LDS 1
#endif
#endif
#ifndef USE_ASYNC_LDS
#define USE_ASYNC_LDS 0
#endif

static __device__ __forceinline__ void copy16_to_lds(const __bf16* g, __bf16* l) {
#if USE_ASYNC_LDS
    v4i* gg = (v4i*)g;                 // reinterpret pointee (generic AS)
    v4i* lg = (v4i*)l;
    __builtin_amdgcn_global_load_async_to_lds_b128(
        (v4i_as1*)gg, (v4i_as3*)lg, 0, 0);
#else
    *(v8bf*)l = *(const v8bf*)g;       // fallback: global_load_b128 + ds_store_b128
#endif
}

static __device__ __forceinline__ void async_wait0() {
#if USE_ASYNC_LDS
#if __has_builtin(__builtin_amdgcn_s_wait_asynccnt)
    __builtin_amdgcn_s_wait_asynccnt(0);
#else
    asm volatile("s_wait_asynccnt 0x0" ::: "memory");
#endif
#endif
}

// ---------------------------------------------------------------------------
// WMMA fragment helpers (wave32; contiguous k order, same for A and B)
// A 16x32 bf16: row=lane&15, elems j: K=16h+j
// ---------------------------------------------------------------------------
static __device__ __forceinline__ v16bf load_a_rm(const __bf16* p, int ld, int lane) {
    int row = lane & 15;
    int h   = lane >> 4;
    const __bf16* r = p + (size_t)row * ld + 16 * h;
    v8bf lo = *(const v8bf*)(r);          // K = 16h .. 16h+7
    v8bf hi = *(const v8bf*)(r + 8);      // K = 16h+8 .. 16h+15
    return __builtin_shufflevector(lo, hi, 0,1,2,3,4,5,6,7,8,9,10,11,12,13,14,15);
}

// B 32x16 bf16 where the K dimension is contiguous in memory (B^T row-major):
// col n = lane&15, elems j: K = 16*(lane>>4) + j  -> one 32B load
static __device__ __forceinline__ v16bf load_b_kc(const __bf16* p, int ld, int lane) {
    int n = lane & 15;
    int h = lane >> 4;
    return *(const v16bf*)(p + (size_t)n * ld + 16 * h);
}

static __device__ __forceinline__ v8f wmma_bf16(v16bf a, v16bf b, v8f c) {
    return __builtin_amdgcn_wmma_f32_16x16x32_bf16(false, a, false, b, (short)0, c,
                                                   false, false);
}

// ---------------------------------------------------------------------------
// Elementwise conversion kernels
// ---------------------------------------------------------------------------
__global__ __launch_bounds__(256) void k_f32_to_bf16(const float* in, __bf16* out, int n) {
    int i = blockIdx.x * 256 + threadIdx.x;
    if (i < n) out[i] = (__bf16)in[i];
}

// weights stored (K,N) f32 -> bf16 transposed (N,K) so GEMM B loads are contiguous
__global__ __launch_bounds__(256) void k_f32_to_bf16_T(const float* in, __bf16* out,
                                                       int K, int N) {
    int i = blockIdx.x * 256 + threadIdx.x;
    if (i >= K * N) return;
    int k = i / N, n = i % N;
    out[(size_t)n * K + k] = (__bf16)in[i];
}

// V (B*T, D) bf16 -> Vt [b, d, t] bf16 so attention P@V B-fragments are contiguous
__global__ __launch_bounds__(256) void k_transpose_v(const __bf16* v, __bf16* vt) {
    int i = blockIdx.x * 256 + threadIdx.x;
    if (i >= BB * TT * DDIM) return;
    int d = i % DDIM;
    int t = (i / DDIM) % TT;
    int b = i / (DDIM * TT);
    vt[(size_t)b * DDIM * TT + (size_t)d * TT + t] = v[i];
}

// ---------------------------------------------------------------------------
// Temporal-bias MLP:  bias[b,i,j] = W2 . leaky_relu( (1/log(e+dt)) * W1 + b1 ) + b2
// Head-independent -> compute once, 16.8MB fits in L2 for the H=16 rereads.
// ---------------------------------------------------------------------------
__global__ __launch_bounds__(256) void k_bias_mlp(const float* __restrict__ dt,
                                                  const float* __restrict__ w1,
                                                  const float* __restrict__ b1,
                                                  const float* __restrict__ w2,
                                                  const float* __restrict__ b2,
                                                  float* __restrict__ bias, int n) {
    int i = blockIdx.x * 256 + threadIdx.x;
    if (i >= n) return;
    float tm  = 1.0f / logf(2.7182818284590452f + dt[i]);
    float acc = b2[0];
#pragma unroll 8
    for (int j = 0; j < TBD; ++j) {
        float h = fmaf(tm, w1[j], b1[j]);
        h = (h > 0.0f) ? h : 0.2f * h;
        acc = fmaf(h, w2[j], acc);
    }
    bias[i] = acc;
}

// ---------------------------------------------------------------------------
// bf16 WMMA GEMM:  C(M,N) = A(M,K) @ B(K,N) + biasvec
// A row-major bf16, Bt = B transposed (N,K) bf16.
// Block: 256 threads = 8 waves, tile 128x64 (wave = 16 rows x 64 cols).
// The shared 64x32 B tile is double-buffered in LDS via async global->LDS DMA
// (ASYNCcnt) so WMMA issue overlaps the next tile's copy.
// mode 0: bf16 output, mode 1: f32 output.
// ---------------------------------------------------------------------------
__global__ __launch_bounds__(256) void k_gemm_bf16(const __bf16* __restrict__ A,
                                                   const __bf16* __restrict__ Bt,
                                                   const float* __restrict__ biasvec,
                                                   __bf16* __restrict__ outb,
                                                   float* __restrict__ outf,
                                                   int M, int N, int K, int mode) {
    __shared__ __bf16 btile[2][64 * 32];          // 2 x 4KB
    const int tid  = threadIdx.x;
    const int lane = tid & 31;
    const int wave = tid >> 5;
    const int row0 = blockIdx.y * 128 + wave * 16;
    const int col0 = blockIdx.x * 64;

    // staging: 256 threads x 16B = 4KB tile; thread -> (row 0..63, 8-elem seg 0..3)
    const int srow = tid >> 2;
    const int sseg = (tid & 3) * 8;
    const __bf16* Bg = Bt + (size_t)(col0 + srow) * K + sseg;
    __bf16*       Bl0 = &btile[0][srow * 32 + sseg];
    __bf16*       Bl1 = &btile[1][srow * 32 + sseg];

    copy16_to_lds(Bg, Bl0);                        // prologue: tile for k=0

    v8f acc[4] = {{}, {}, {}, {}};
    const __bf16* Arow = A + (size_t)row0 * K;

    for (int k = 0; k < K; k += 32) {
        const int cur = (k >> 5) & 1;
        async_wait0();
        __syncthreads();                           // LDS tile 'cur' ready for all waves
        if (k + 32 < K)                            // overlap next DMA with WMMAs
            copy16_to_lds(Bg + k + 32, cur ? Bl0 : Bl1);

        __builtin_prefetch(Arow + k + 128, 0, 1);  // global_prefetch_b8
        v16bf a = load_a_rm(Arow + k, K, lane);
        const __bf16* bt = &btile[cur][0];
#pragma unroll
        for (int c = 0; c < 4; ++c) {
            v16bf b = load_b_kc(bt + (c * 16) * 32, 32, lane);  // ds_load_b128 pairs
            acc[c] = wmma_bf16(a, b, acc[c]);
        }
    }

    int nl = lane & 15, hh = lane >> 4;
#pragma unroll
    for (int c = 0; c < 4; ++c) {
        int col = col0 + c * 16 + nl;
        float bv = biasvec ? biasvec[col] : 0.0f;
#pragma unroll
        for (int v = 0; v < 8; ++v) {
            int row = row0 + v + 8 * hh;
            float val = acc[c][v] + bv;
            if (mode == 0) outb[(size_t)row * N + col] = (__bf16)val;
            else           outf[(size_t)row * N + col] = val;
        }
    }
}

// ---------------------------------------------------------------------------
// Flash attention with temporal bias + padding + causal(CLS row free) masks.
// One wave (32 threads) per (b, head, 16-row q tile). Online softmax,
// K stepped 32 keys/iter: 4 WMMAs for S, 4 WMMAs for P@V. P re-layout via LDS.
// ---------------------------------------------------------------------------
__global__ __launch_bounds__(32) void k_attention(const __bf16* __restrict__ qb,
                                                  const __bf16* __restrict__ kb,
                                                  const __bf16* __restrict__ vtb,
                                                  const float* __restrict__ bias,
                                                  const unsigned char* __restrict__ pad,
                                                  __bf16* __restrict__ ob) {
    __shared__ __bf16 plds[16 * 32];
    const int lane = threadIdx.x;
    const int blk  = blockIdx.x;          // B*H*(T/16) = 4096
    const int qt   = blk & 63;
    const int head = (blk >> 6) & (HNUM - 1);
    const int b    = blk >> 10;
    const int q0   = qt * 16;
    const int nl   = lane & 15, hh = lane >> 4;
    const float scale = 0.125f;           // DK^-0.5

    const __bf16* Q = qb + ((size_t)(b * TT + q0)) * DDIM + head * DK;
    v16bf qa0 = load_a_rm(Q, DDIM, lane);
    v16bf qa1 = load_a_rm(Q + 32, DDIM, lane);

    v8f   o[4] = {{}, {}, {}, {}};
    float m_i[8], l_i[8];
#pragma unroll
    for (int v = 0; v < 8; ++v) { m_i[v] = -3.0e38f; l_i[v] = 0.0f; }

    const size_t bmat0 = (size_t)b * TT * TT;

    for (int k0 = 0; k0 < TT; k0 += 32) {
        // ---- S = Q @ K^T for 32 keys (two 16x16 tiles) ----
        v8f s[2];
#pragma unroll
        for (int t2 = 0; t2 < 2; ++t2) {
            const __bf16* Kp = kb + ((size_t)(b * TT + k0 + t2 * 16)) * DDIM + head * DK;
            v8f st = {};
            st = wmma_bf16(qa0, load_b_kc(Kp, DDIM, lane), st);
            st = wmma_bf16(qa1, load_b_kc(Kp + 32, DDIM, lane), st);
            s[t2] = st;
        }
        // ---- scale + temporal bias + masks, track row max ----
        float rowmax[8];
#pragma unroll
        for (int v = 0; v < 8; ++v) rowmax[v] = -3.0e38f;
#pragma unroll
        for (int t2 = 0; t2 < 2; ++t2) {
            int key = k0 + t2 * 16 + nl;
#pragma unroll
            for (int v = 0; v < 8; ++v) {
                int qrow = q0 + v + 8 * hh;
                size_t mi = bmat0 + (size_t)qrow * TT + key;
                float val = fmaf(s[t2][v], scale, bias[mi]);
                bool ok = (pad[mi] != 0) && !((key > qrow) && (qrow != 0));
                val = ok ? val : -3.0e38f;
                s[t2][v]  = val;
                rowmax[v] = fmaxf(rowmax[v], val);
            }
        }
#pragma unroll
        for (int v = 0; v < 8; ++v)
#pragma unroll
            for (int msk = 1; msk < 16; msk <<= 1)
                rowmax[v] = fmaxf(rowmax[v], __shfl_xor(rowmax[v], msk, 32));

        // ---- online softmax update ----
#pragma unroll
        for (int v = 0; v < 8; ++v) {
            float mn   = fmaxf(m_i[v], rowmax[v]);
            float corr = __expf(m_i[v] - mn);
            m_i[v] = mn;
            float sum = 0.0f;
#pragma unroll
            for (int t2 = 0; t2 < 2; ++t2) {
                float p = __expf(s[t2][v] - mn);
                s[t2][v] = p;
                sum += p;
            }
#pragma unroll
            for (int msk = 1; msk < 16; msk <<= 1)
                sum += __shfl_xor(sum, msk, 32);
            l_i[v] = l_i[v] * corr + sum;
#pragma unroll
            for (int c = 0; c < 4; ++c) o[c][v] *= corr;
        }

        // ---- re-layout P (C layout -> A layout) through LDS ----
        // Same-wave LDS ops are in-order; one barrier (waitcnt fence) suffices.
#pragma unroll
        for (int t2 = 0; t2 < 2; ++t2)
#pragma unroll
            for (int v = 0; v < 8; ++v)
                plds[(v + 8 * hh) * 32 + t2 * 16 + nl] = (__bf16)s[t2][v];
        __syncthreads();   // single-wave WG: lowers to s_wait_dscnt + s_nop
        v16bf pa = load_a_rm(plds, 32, lane);

        // ---- O += P @ V  (V pre-transposed: [b, d, t]) ----
#pragma unroll
        for (int c = 0; c < 4; ++c) {
            const __bf16* Vp = vtb + (size_t)b * DDIM * TT +
                               (size_t)(head * DK + c * 16) * TT + k0;
            o[c] = wmma_bf16(pa, load_b_kc(Vp, TT, lane), o[c]);
        }
    }

    // ---- finalize: O / l, store bf16 per-head columns ----
    float invl[8];
#pragma unroll
    for (int v = 0; v < 8; ++v) invl[v] = 1.0f / l_i[v];
#pragma unroll
    for (int c = 0; c < 4; ++c)
#pragma unroll
        for (int v = 0; v < 8; ++v) {
            int row = q0 + v + 8 * hh;
            int col = head * DK + c * 16 + nl;
            ob[((size_t)(b * TT + row)) * DDIM + col] = (__bf16)(o[c][v] * invl[v]);
        }
}

// ---------------------------------------------------------------------------
// Host-side orchestration
// ---------------------------------------------------------------------------
extern "C" void kernel_launch(void* const* d_in, const int* in_sizes, int n_in,
                              void* d_out, int out_size, void* d_ws, size_t ws_size,
                              hipStream_t stream) {
    const float* x  = (const float*)d_in[0];
    const unsigned char* pad = (const unsigned char*)d_in[1];
    const float* dt = (const float*)d_in[2];
    const float* wq = (const float*)d_in[3];  const float* bq = (const float*)d_in[4];
    const float* wk = (const float*)d_in[5];  const float* bk = (const float*)d_in[6];
    const float* wv = (const float*)d_in[7];  const float* bv = (const float*)d_in[8];
    const float* wo = (const float*)d_in[9];  const float* bo = (const float*)d_in[10];
    const float* w1 = (const float*)d_in[11]; const float* b1 = (const float*)d_in[12];
    const float* w2 = (const float*)d_in[13]; const float* b2 = (const float*)d_in[14];
    float* out = (float*)d_out;

    char* ws = (char*)d_ws;
    size_t off = 0;
    auto alloc = [&](size_t bytes) -> void* {
        void* p = ws + off;
        off += (bytes + 255) & ~(size_t)255;
        return p;
    };
    const size_t NTOK = (size_t)BB * TT;            // 4096
    __bf16* xb    = (__bf16*)alloc(NTOK * DDIM * 2);
    __bf16* wqT   = (__bf16*)alloc((size_t)DDIM * DDIM * 2);
    __bf16* wkT   = (__bf16*)alloc((size_t)DDIM * DDIM * 2);
    __bf16* wvT   = (__bf16*)alloc((size_t)DDIM * DDIM * 2);
    __bf16* woT   = (__bf16*)alloc((size_t)DDIM * DDIM * 2);
    __bf16* qb    = (__bf16*)alloc(NTOK * DDIM * 2);
    __bf16* kb    = (__bf16*)alloc(NTOK * DDIM * 2);
    __bf16* vb    = (__bf16*)alloc(NTOK * DDIM * 2);
    __bf16* vtb   = (__bf16*)alloc(NTOK * DDIM * 2);
    __bf16* attnb = (__bf16*)alloc(NTOK * DDIM * 2);
    float*  biasM = (float*)alloc((size_t)BB * TT * TT * 4);

    const int nX  = BB * TT * DDIM;                 // 4,194,304
    const int nW  = DDIM * DDIM;                    // 1,048,576
    const int nBM = BB * TT * TT;                   // 4,194,304

    k_f32_to_bf16<<<(nX + 255) / 256, 256, 0, stream>>>(x, xb, nX);
    k_f32_to_bf16_T<<<(nW + 255) / 256, 256, 0, stream>>>(wq, wqT, DDIM, DDIM);
    k_f32_to_bf16_T<<<(nW + 255) / 256, 256, 0, stream>>>(wk, wkT, DDIM, DDIM);
    k_f32_to_bf16_T<<<(nW + 255) / 256, 256, 0, stream>>>(wv, wvT, DDIM, DDIM);
    k_f32_to_bf16_T<<<(nW + 255) / 256, 256, 0, stream>>>(wo, woT, DDIM, DDIM);
    k_bias_mlp<<<(nBM + 255) / 256, 256, 0, stream>>>(dt, w1, b1, w2, b2, biasM, nBM);

    dim3 ggrid(DDIM / 64, (int)(NTOK / 128));       // 16 x 32
    k_gemm_bf16<<<ggrid, 256, 0, stream>>>(xb, wqT, bq, qb, nullptr,
                                           (int)NTOK, DDIM, DDIM, 0);
    k_gemm_bf16<<<ggrid, 256, 0, stream>>>(xb, wkT, bk, kb, nullptr,
                                           (int)NTOK, DDIM, DDIM, 0);
    k_gemm_bf16<<<ggrid, 256, 0, stream>>>(xb, wvT, bv, vb, nullptr,
                                           (int)NTOK, DDIM, DDIM, 0);
    k_transpose_v<<<(nX + 255) / 256, 256, 0, stream>>>(vb, vtb);

    k_attention<<<BB * HNUM * (TT / 16), 32, 0, stream>>>(qb, kb, vtb, biasM, pad, attnb);

    k_gemm_bf16<<<ggrid, 256, 0, stream>>>(attnb, woT, bo, nullptr, out,
                                           (int)NTOK, DDIM, DDIM, 1);
    (void)in_sizes; (void)n_in; (void)out_size; (void)ws_size;
}
